// WindowAttention_71949292143063
// MI455X (gfx1250) — matrix operand order repaired and checked by reading, not compiled, not run on hardware
//
#include <hip/hip_runtime.h>
#include <hip/hip_bf16.h>
#include <cstddef>

// ---------------------------------------------------------------------------
// Window attention, fully fused per-window, CDNA5 (gfx1250) WMMA f16 path.
//   HW=49 (padded to 64), C=128, H=4 heads, HD=32, BS=16384, NUM_WIN=1024
// ---------------------------------------------------------------------------

typedef __attribute__((ext_vector_type(16))) _Float16 v16h;
typedef __attribute__((ext_vector_type(8)))  _Float16 v8h;
typedef __attribute__((ext_vector_type(8)))  float    v8f;

#define HWIN   49
#define CDIM   128
#define NHEAD  4
#define HDIM   32
#define PADR   64
#define NWIN   1024
#define SCALE  0.17677669529663687f   /* 32^-0.5 */

// ---- WMMA wrapper: D = A(16x32 f16) * B(32x16 f16) + C(16x16 f32) ----------
static __device__ __forceinline__ v8f wmma16(v16h a, v16h b, v8f c) {
    return __builtin_amdgcn_wmma_f32_16x16x32_f16(
        /*neg_a=*/false, a, /*neg_b=*/false, b,
        /*c_mod=*/(short)0, c, /*reuse_a=*/false, /*reuse_b=*/false);
}

// ---- A-fragment load (16x32 f16, contiguous-k layout, k = 16*lh + i) ------------------------
// lane = lh*16 + lm ; a[i] = A[m][kbase + lh*16 + i]
static __device__ __forceinline__ v16h load_afrag(const _Float16* base, int stride,
                                                  int m, int kbase, int lh) {
    const _Float16* p = base + m * stride + kbase + lh * 16;
    v8h lo = *(const v8h*)(p);
    v8h hi = *(const v8h*)(p + 8);
    v16h r;
#pragma unroll
    for (int i = 0; i < 8; ++i) { r[i] = lo[i]; r[i + 8] = hi[i]; }
    return r;
}

// ---- B-fragment load: b[i] = B[kbase + lh*16 + i][n] with B stored so that
// the 16 K-consecutive elements for column n are contiguous at `p`. ----------
static __device__ __forceinline__ v16h load_bfrag(const _Float16* p) {
    return *(const v16h*)(p);
}

// ===========================================================================
// Prep: f32 -> f16 weights, and padded [4][64][64] relative-position bias.
// ===========================================================================
__global__ void prep_kernel(const float* __restrict__ qkv_w,
                            const float* __restrict__ proj_w,
                            const float* __restrict__ bias_table,
                            const int*   __restrict__ rel_index,
                            _Float16* __restrict__ qkv_wh,
                            _Float16* __restrict__ proj_wh,
                            float* __restrict__ biasm) {
    int i = blockIdx.x * blockDim.x + threadIdx.x;
    if (i < 3 * CDIM * CDIM) {                       // 49152
        qkv_wh[i] = (_Float16)qkv_w[i];
    } else if (i < 3 * CDIM * CDIM + CDIM * CDIM) {  // +16384
        int j = i - 3 * CDIM * CDIM;
        proj_wh[j] = (_Float16)proj_w[j];
    } else if (i < 3 * CDIM * CDIM + CDIM * CDIM + NHEAD * PADR * PADR) {
        int j = i - (3 * CDIM * CDIM + CDIM * CDIM);
        int h = j >> 12, q = (j >> 6) & 63, k = j & 63;
        float v = 0.0f;
        if (q < HWIN && k < HWIN)
            v = bias_table[rel_index[q * HWIN + k] * NHEAD + h];
        biasm[j] = v;
    }
}

// ===========================================================================
// Fused window attention: one 128-thread block per window; wave w == head w.
// ===========================================================================
__global__ __launch_bounds__(128)
void win_attn_kernel(const float* __restrict__ x,
                     const float* __restrict__ mask,
                     const float* __restrict__ qkv_b,
                     const float* __restrict__ proj_b,
                     const _Float16* __restrict__ qkv_wh,
                     const _Float16* __restrict__ proj_wh,
                     const float* __restrict__ biasm,
                     float* __restrict__ out) {
    const int w    = blockIdx.x;
    const int tid  = threadIdx.x;
    const int head = tid >> 5;          // wave id == head
    const int lane = tid & 31;
    const int lm   = lane & 15;         // column / M within fragment
    const int lh   = lane >> 4;         // lane half

    __shared__ __align__(32) _Float16 q_s [NHEAD][PADR][HDIM];  // 16 KB (reused as attn_out)
    __shared__ __align__(32) _Float16 k_s [NHEAD][PADR][HDIM];  // 16 KB
    __shared__ __align__(32) _Float16 vT_s[NHEAD][HDIM][PADR];  // 16 KB
    __shared__ __align__(32) _Float16 p_s [NHEAD][16][PADR];    //  8 KB
    __shared__ __align__(32) _Float16 x_s [16][CDIM];           //  4 KB

    const float* xw = x + (size_t)w * HWIN * CDIM;

    // ------------------------------ QKV ------------------------------------
#pragma unroll 1
    for (int mt = 0; mt < 4; ++mt) {
        __syncthreads();
        {   // stage 16 rows of x as f16 (zero-pad rows >= 49)
            int row = tid >> 3;
            int c0  = (tid & 7) * 16;
            int gr  = mt * 16 + row;
            _Float16* dst = &x_s[row][c0];
            if (gr < HWIN) {
                const float4* src = (const float4*)(xw + (size_t)gr * CDIM + c0);
                float4 f[4] = {src[0], src[1], src[2], src[3]};
#pragma unroll
                for (int u = 0; u < 4; ++u) {
                    dst[4 * u + 0] = (_Float16)f[u].x;
                    dst[4 * u + 1] = (_Float16)f[u].y;
                    dst[4 * u + 2] = (_Float16)f[u].z;
                    dst[4 * u + 3] = (_Float16)f[u].w;
                }
            } else {
#pragma unroll
                for (int u = 0; u < 16; ++u) dst[u] = (_Float16)0.0f;
            }
        }
        __syncthreads();

        v16h af[4];
#pragma unroll
        for (int ks = 0; ks < 4; ++ks)
            af[ks] = load_afrag(&x_s[0][0], CDIM, lm, 32 * ks, lh);

#pragma unroll
        for (int part = 0; part < 6; ++part) {
            const int sel  = part >> 1;                 // 0=q 1=k 2=v
            const int j    = part & 1;                  // 16-col half of head dim
            const int ncol = sel * CDIM + head * HDIM + j * 16;
            v8f acc = {};
#pragma unroll
            for (int ks = 0; ks < 4; ++ks) {
                v16h bf = load_bfrag(qkv_wh + (size_t)(ncol + lm) * CDIM
                                            + 32 * ks + 16 * lh);
                acc = wmma16(af[ks], bf, acc);
            }
            const float bcol = qkv_b[ncol + lm];
#pragma unroll
            for (int r = 0; r < 8; ++r) {
                int row = mt * 16 + 8 * lh + r;
                float val = acc[r] + bcol;
                if (sel == 0)      q_s [head][row][j * 16 + lm] = (_Float16)(val * SCALE);
                else if (sel == 1) k_s [head][row][j * 16 + lm] = (_Float16)val;
                else               vT_s[head][j * 16 + lm][row] = (_Float16)val;
            }
        }
    }
    // Each wave consumes only its own head's q/k/v -> no barrier needed here.

    // --------------------------- Attention ---------------------------------
    const float* maskw = mask  + (size_t)(w & (NWIN - 1)) * HWIN * HWIN;
    const float* bm    = biasm + (size_t)head * PADR * PADR;

#pragma unroll 1
    for (int mt = 0; mt < 4; ++mt) {
        // S = q @ k^T : 4 tiles along keys, K = head_dim = 32
        v16h qa = load_afrag(&q_s[head][0][0], HDIM, 16 * mt + lm, 0, lh);
        v8f s[4];
#pragma unroll
        for (int nt = 0; nt < 4; ++nt) {
            v16h kb = load_bfrag(&k_s[head][16 * nt + lm][16 * lh]);
            v8f z = {};
            s[nt] = wmma16(qa, kb, z);
        }

        // softmax (rows 16mt+8*lh+r, row spread over 16 lanes x 4 tiles)
#pragma unroll
        for (int r = 0; r < 8; ++r) {
            const int qrow = 16 * mt + 8 * lh + r;
            float tv[4];
            float mx = -3.0e38f;
#pragma unroll
            for (int nt = 0; nt < 4; ++nt) {
                int kc = 16 * nt + lm;
                float val = s[nt][r] + bm[qrow * PADR + kc];
                if (kc < HWIN) {
                    if (qrow < HWIN) val += maskw[qrow * HWIN + kc];
                } else {
                    val = -1.0e30f;
                }
                tv[nt] = val;
                mx = fmaxf(mx, val);
            }
#pragma unroll
            for (int d = 1; d < 16; d <<= 1)
                mx = fmaxf(mx, __shfl_xor(mx, d, 32));
            float sum = 0.0f;
#pragma unroll
            for (int nt = 0; nt < 4; ++nt) {
                float e = __expf(tv[nt] - mx);
                tv[nt] = e;
                sum += e;
            }
#pragma unroll
            for (int d = 1; d < 16; d <<= 1)
                sum += __shfl_xor(sum, d, 32);
            const float inv = 1.0f / sum;
#pragma unroll
            for (int nt = 0; nt < 4; ++nt)
                p_s[head][8 * lh + r][16 * nt + lm] = (_Float16)(tv[nt] * inv);
        }

        // O = P @ v  (K = 64 over two 32-steps); write O back into q_s region
#pragma unroll
        for (int nt2 = 0; nt2 < 2; ++nt2) {
            v8f oacc = {};
#pragma unroll
            for (int ks = 0; ks < 2; ++ks) {
                v16h pa = load_afrag(&p_s[head][0][0], PADR, lm, 32 * ks, lh);
                v16h vb = load_bfrag(&vT_s[head][16 * nt2 + lm][32 * ks + 16 * lh]);
                oacc = wmma16(pa, vb, oacc);
            }
#pragma unroll
            for (int r = 0; r < 8; ++r)
                q_s[head][16 * mt + 8 * lh + r][16 * nt2 + lm] = (_Float16)oacc[r];
        }
    }
    __syncthreads();   // proj reads all heads' attn_out from q_s

    // ----------------------------- Proj ------------------------------------
    // attn_out[row][c] == q_s[c>>5][row][c&31]; each wave does 2 N-tiles x 4 M-tiles
#pragma unroll 1
    for (int t8 = 0; t8 < 8; ++t8) {
        const int mt   = t8 >> 1;
        const int jn   = t8 & 1;
        const int ncol = head * 32 + jn * 16;
        v8f acc = {};
#pragma unroll
        for (int ks = 0; ks < 4; ++ks) {   // K block 32*ks..32*ks+31 lives in head `ks`
            v16h pa = load_afrag(&q_s[ks][0][0], HDIM, 16 * mt + lm, 0, lh);
            v16h pb = load_bfrag(proj_wh + (size_t)(ncol + lm) * CDIM
                                         + 32 * ks + 16 * lh);
            acc = wmma16(pa, pb, acc);
        }
        const float bc = proj_b[ncol + lm];
#pragma unroll
        for (int r = 0; r < 8; ++r) {
            int row = 16 * mt + 8 * lh + r;
            if (row < HWIN)
                out[((size_t)w * HWIN + row) * CDIM + ncol + lm] = acc[r] + bc;
        }
    }
}

// ===========================================================================
extern "C" void kernel_launch(void* const* d_in, const int* in_sizes, int n_in,
                              void* d_out, int out_size, void* d_ws, size_t ws_size,
                              hipStream_t stream) {
    const float* x          = (const float*)d_in[0];
    const float* mask       = (const float*)d_in[1];
    const float* qkv_w      = (const float*)d_in[2];
    const float* qkv_b      = (const float*)d_in[3];
    const float* proj_w     = (const float*)d_in[4];
    const float* proj_b     = (const float*)d_in[5];
    const float* bias_table = (const float*)d_in[6];
    const int*   rel_index  = (const int*)d_in[7];

    char* ws = (char*)d_ws;
    _Float16* qkv_wh  = (_Float16*)(ws);                      // 384*128*2  = 98304 B
    _Float16* proj_wh = (_Float16*)(ws + 98304);              // 128*128*2  = 32768 B
    float*    biasm   = (float*)   (ws + 131072);             // 4*64*64*4  = 65536 B
                                                              // total ws   = 196608 B

    const int prep_total = 3 * CDIM * CDIM + CDIM * CDIM + NHEAD * PADR * PADR;
    prep_kernel<<<(prep_total + 255) / 256, 256, 0, stream>>>(
        qkv_w, proj_w, bias_table, rel_index, qkv_wh, proj_wh, biasm);

    win_attn_kernel<<<16384, 128, 0, stream>>>(
        x, mask, qkv_b, proj_b, qkv_wh, proj_wh, biasm, (float*)d_out);
}
